// ContrastiveLoss_83391085019222
// MI455X (gfx1250) — matrix operand order repaired and checked by reading, compiled, not run on hardware
//
#include <hip/hip_runtime.h>
#include <math.h>

// Problem constants (match reference)
#define GN   4096            // N
#define LDC  4097            // N+1 (row stride of S = [sims | sims_no])
#define GD   512             // D
#define NCNUM 409            // int(0.1 * N)
#define CHUNKS 32
#define ROWSPC (GN / CHUNKS) // 128 rows per column-pass chunk

static __device__ __constant__ const float kREG    = 0.03f;
#define kINVREG (1.0f / 0.03f)
#define kGAMMA  0.8f
#define kTHRESH 1e-4f

typedef __attribute__((ext_vector_type(16))) __bf16 v16bf;
typedef __attribute__((ext_vector_type(8)))  __bf16 v8bf;
typedef __attribute__((ext_vector_type(8)))  float  v8f;

// ---------------- helpers ----------------
__device__ __forceinline__ void lse_add(float& m, float& s, float a) {
  if (a <= m) { s += __expf(a - m); }
  else        { s = s * __expf(m - a) + 1.0f; m = a; }
}
__device__ __forceinline__ void lse_merge(float& m, float& s, float m2, float s2) {
  if (m2 == -INFINITY) return;
  if (m  == -INFINITY) { m = m2; s = s2; return; }
  if (m2 <= m) s += s2 * __expf(m2 - m);
  else { s = s * __expf(m - m2) + s2; m = m2; }
}
// Sinkhorn mask Mv(i,j): ones; last col 0; except NC rows: (nc,nc)=0, (nc,last)=1
__device__ __forceinline__ bool mask_ij(int i, int j, int nci) {
  return (j == GN) ? (nci != 0) : !((i == j) && (nci != 0));
}

// ---------------- init ----------------
__global__ void zero_kernel(int* p, size_t n) {
  for (size_t i = (size_t)blockIdx.x * blockDim.x + threadIdx.x; i < n;
       i += (size_t)gridDim.x * blockDim.x) p[i] = 0;
}
__global__ void setmin_kernel(unsigned* p) { *p = 0xFFFFFFFFu; }

// fp32 -> (bf16 hi, bf16 lo) split for near-fp32 WMMA accuracy
__global__ __launch_bounds__(256) void cvt_split_kernel(const float* __restrict__ X,
                                                        __bf16* __restrict__ hi,
                                                        __bf16* __restrict__ lo, int total) {
  int i = blockIdx.x * 256 + threadIdx.x;
  if (i < total) {
    float f = X[i];
    __bf16 h = (__bf16)f;
    hi[i] = h;
    lo[i] = (__bf16)(f - (float)h);
  }
}

// ---------------- GEMM: S[:, :N] = A @ B^T via split-bf16 WMMA ----------------
// block tile 64(i) x 32(j), 8 waves, each wave one 16x16 WMMA tile, K-step 32.
__global__ __launch_bounds__(256) void gemm_wmma_kernel(
    const __bf16* __restrict__ Ahi, const __bf16* __restrict__ Alo,
    const __bf16* __restrict__ Bhi, const __bf16* __restrict__ Blo,
    float* __restrict__ S) {
  __shared__ __align__(16) __bf16 sAh[64 * 32];
  __shared__ __align__(16) __bf16 sAl[64 * 32];
  __shared__ __align__(16) __bf16 sBh[32 * 32];
  __shared__ __align__(16) __bf16 sBl[32 * 32];

  const int i0 = blockIdx.y * 64;
  const int j0 = blockIdx.x * 32;
  const int t = threadIdx.x;
  const int wave = t >> 5, lane = t & 31;
  const int twi = wave & 3;   // 0..3 -> i offset
  const int twj = wave >> 2;  // 0..1 -> j offset
  const int half = lane >> 4; // 0/1
  const int l15 = lane & 15;

  // branchless staging assignments (uniform control flow; no exec-mask branches)
  const int rA = t >> 2, kA = (t & 3) * 8;                 // A: 64x32, one v8 per thread
  const int tb = t & 127, rB = tb >> 2, kB = (tb & 3) * 8; // B: 32x32, 128 v8 per matrix
  const __bf16* gB = (t < 128) ? Bhi : Blo;
  __bf16*       sB = (t < 128) ? sBh : sBl;

  v8f acc = {};

  for (int k0 = 0; k0 < GD; k0 += 32) {
    __syncthreads();
    *(v8bf*)&sAh[rA * 32 + kA] = *(const v8bf*)&Ahi[(size_t)(i0 + rA) * GD + k0 + kA];
    *(v8bf*)&sAl[rA * 32 + kA] = *(const v8bf*)&Alo[(size_t)(i0 + rA) * GD + k0 + kA];
    *(v8bf*)&sB [rB * 32 + kB] = *(const v8bf*)&gB [(size_t)(j0 + rB) * GD + k0 + kB];
    __syncthreads();

    // A fragment (16x32, row M = lane&15): lanes<16 hold K {0..7,16..23}, lanes>=16 {8..15,24..31}
    const int ra = (twi * 16 + l15) * 32 + half * 8;
    v8bf a0h = *(const v8bf*)&sAh[ra];
    v8bf a1h = *(const v8bf*)&sAh[ra + 16];
    v8bf a0l = *(const v8bf*)&sAl[ra];
    v8bf a1l = *(const v8bf*)&sAl[ra + 16];
    v16bf ah, al;
#pragma unroll
    for (int e = 0; e < 8; ++e) {
      ah[e] = a0h[e]; ah[8 + e] = a1h[e];
      al[e] = a0l[e]; al[8 + e] = a1l[e];
    }
    // B fragment (32x16, col N = lane&15): lanes<16 hold K {0..7,16..23}, lanes>=16 {8..15,24..31}
    const int rb = (twj * 16 + l15) * 32 + half * 8;
    v8bf b0h = *(const v8bf*)&sBh[rb];
    v8bf b1h = *(const v8bf*)&sBh[rb + 16];
    v8bf b0l = *(const v8bf*)&sBl[rb];
    v8bf b1l = *(const v8bf*)&sBl[rb + 16];
    v16bf bh, bl;
#pragma unroll
    for (int e = 0; e < 8; ++e) {
      bh[e] = b0h[e]; bh[8 + e] = b1h[e];
      bl[e] = b0l[e]; bl[8 + e] = b1l[e];
    }

    acc = __builtin_amdgcn_wmma_f32_16x16x32_bf16(false, ah, false, bh, (short)0, acc, false, false);
    acc = __builtin_amdgcn_wmma_f32_16x16x32_bf16(false, ah, false, bl, (short)0, acc, false, false);
    acc = __builtin_amdgcn_wmma_f32_16x16x32_bf16(false, al, false, bh, (short)0, acc, false, false);
  }

  // C/D layout: VGPR r, lanes<16 -> (M=r, N=lane), lanes>=16 -> (M=8+r, N=lane-16)
  const int gi = i0 + twi * 16 + half * 8;
  const int gj = j0 + twj * 16 + l15;
#pragma unroll
  for (int r = 0; r < 8; ++r) S[(size_t)(gi + r) * LDC + gj] = acc[r];
}

// ---------------- per-row dot: S[:, N] = rowdot(image, text_no) ----------------
__global__ __launch_bounds__(256) void rowdot_kernel(const float* __restrict__ A,
                                                     const float* __restrict__ B,
                                                     float* __restrict__ S) {
  int wave = threadIdx.x >> 5, lane = threadIdx.x & 31;
  int row = blockIdx.x * 8 + wave;
  const float* a = A + (size_t)row * GD;
  const float* b = B + (size_t)row * GD;
  float s = 0.f;
  for (int k = lane; k < GD; k += 32) s += a[k] * b[k];
#pragma unroll
  for (int o = 16; o; o >>= 1) s += __shfl_xor(s, o, 32);
  if (lane == 0) S[(size_t)row * LDC + GN] = s;
}

// ---------------- loss_op: mean(relu(cos+0.2)+relu(-0.7-cos)) ----------------
__global__ __launch_bounds__(256) void loss_op_kernel(const float* __restrict__ T,
                                                      const float* __restrict__ Tn,
                                                      float* acc) {
  int wave = threadIdx.x >> 5, lane = threadIdx.x & 31;
  int row = blockIdx.x * 8 + wave;
  const float* a = T + (size_t)row * GD;
  const float* b = Tn + (size_t)row * GD;
  float s = 0.f;
  for (int k = lane; k < GD; k += 32) s += a[k] * b[k];
#pragma unroll
  for (int o = 16; o; o >>= 1) s += __shfl_xor(s, o, 32);
  if (lane == 0) atomicAdd(acc, fmaxf(s + 0.2f, 0.f) + fmaxf(-0.7f - s, 0.f));
}

// ---------------- global min of S (for cost normalization) ----------------
__global__ __launch_bounds__(256) void gmin_kernel(const float* __restrict__ S, unsigned* minEnc) {
  const size_t total = (size_t)GN * LDC;
  float m = INFINITY;
  for (size_t i = (size_t)blockIdx.x * 256 + threadIdx.x; i < total; i += (size_t)gridDim.x * 256)
    m = fminf(m, S[i]);
  __shared__ float sm[256];
  sm[threadIdx.x] = m; __syncthreads();
  for (int o = 128; o; o >>= 1) { if (threadIdx.x < o) sm[threadIdx.x] = fminf(sm[threadIdx.x], sm[threadIdx.x + o]); __syncthreads(); }
  if (threadIdx.x == 0) {
    unsigned u = __float_as_uint(sm[0]);
    unsigned e = (u >> 31) ? ~u : (u | 0x80000000u);  // order-preserving encode
    atomicMin(minEnc, e);
  }
}

// scal: [0]=minEnc(u32) [1]=invMaxC [2]=scale [3]=err_v [4]=err_t [5]=acc_img [6]=acc_txt [7]=acc_op [8]=conv_v [9]=conv_t
__global__ void prep_kernel(const float* __restrict__ logit_scale, float* scal) {
  unsigned e = ((const unsigned*)scal)[0];
  unsigned u = (e & 0x80000000u) ? (e & 0x7FFFFFFFu) : ~e;
  float minS = __uint_as_float(u);
  scal[1] = 1.0f / (1.0f - minS);   // 1/max(1-S)
  scal[2] = expf(logit_scale[0]);
}

// ---------------- preds (infoNCE diagonal softmax prob) ----------------
__global__ __launch_bounds__(256) void preds_kernel(const float* __restrict__ S,
                                                    const float* __restrict__ scal,
                                                    float* __restrict__ preds) {
  int i = blockIdx.x, tid = threadIdx.x;
  float scale = scal[2];
  const float* row = S + (size_t)i * LDC;
  float m = -INFINITY, s = 0.f;
  for (int j = tid; j < GN; j += 256) {
    __builtin_prefetch(row + j + 1024, 0, 3);  // global_prefetch_b8, 4 iters ahead
    lse_add(m, s, scale * row[j]);
  }
  __shared__ float sm[256], ss[256];
  sm[tid] = m; ss[tid] = s; __syncthreads();
  for (int o = 128; o; o >>= 1) { if (tid < o) lse_merge(sm[tid], ss[tid], sm[tid + o], ss[tid + o]); __syncthreads(); }
  if (tid == 0) preds[i] = __expf(scale * row[i] - (logf(ss[0]) + sm[0]));
}

// ---------------- top-k smallest preds via one-block bitonic sort ----------------
__global__ __launch_bounds__(1024) void topk_kernel(const float* __restrict__ preds, int* __restrict__ isNC) {
  __shared__ float key[GN];
  __shared__ int   idx[GN];
  for (int i = threadIdx.x; i < GN; i += 1024) { key[i] = preds[i]; idx[i] = i; }
  __syncthreads();
  for (int k = 2; k <= GN; k <<= 1)
    for (int j = k >> 1; j > 0; j >>= 1) {
      for (int i = threadIdx.x; i < GN; i += 1024) {
        int ixj = i ^ j;
        if (ixj > i) {
          bool up = ((i & k) == 0);
          float a = key[i], b = key[ixj];
          int ia = idx[i], ib = idx[ixj];
          bool agtb = (a > b) || (a == b && ia > ib);
          if (up == agtb) { key[i] = b; key[ixj] = a; idx[i] = ib; idx[ixj] = ia; }
        }
      }
      __syncthreads();
    }
  for (int t = threadIdx.x; t < NCNUM; t += 1024) isNC[idx[t]] = 1;
}

// ---------------- Sinkhorn: generic row pass (LSE along j of S rows) ----------------
// val(i,j) = ((S[i,j]-1)*invMaxC + a[j] + b[i]) / REG, masked by Mv(i,j).
// b[i] <- REG*(logTarget - lse) + b[i]; optional err accumulation.
__global__ __launch_bounds__(256) void sink_row_kernel(
    const float* __restrict__ S, const int* __restrict__ isNC,
    const float* __restrict__ a, float* __restrict__ b,
    const float* __restrict__ scal, float logTarget, float* errAcc, const int* conv) {
  if (*conv) return;
  const float invMaxC = scal[1];
  const int i = blockIdx.x, tid = threadIdx.x;
  const float bi = b[i];
  const int nci = isNC[i];
  const float* row = S + (size_t)i * LDC;
  float m = -INFINITY, s = 0.f;
  for (int j = tid; j < LDC; j += 256) {
    __builtin_prefetch(row + j + 1024, 0, 3);  // stream the row ahead of the LSE
    if (mask_ij(i, j, nci)) {
      float val = ((row[j] - 1.0f) * invMaxC + a[j] + bi) * kINVREG;
      lse_add(m, s, val);
    }
  }
  __shared__ float sm[256], ss[256];
  sm[tid] = m; ss[tid] = s; __syncthreads();
  for (int o = 128; o; o >>= 1) { if (tid < o) lse_merge(sm[tid], ss[tid], sm[tid + o], ss[tid + o]); __syncthreads(); }
  if (tid == 0) {
    float lse = logf(ss[0] + 1e-10f) + sm[0];
    float bn = kREG * (logTarget - lse) + bi;
    if (errAcc) atomicAdd(errAcc, fabsf(bn - bi));
    b[i] = bn;
  }
}

// ---------------- Sinkhorn: column pass, stage 1 (coalesced: lane-per-column) --------
// val(i,j) = ((S[i,j]-1)*invMaxC + c[i] + d[j]) / REG, masked. Partial (m,s) per (chunk,col).
__global__ __launch_bounds__(256) void sink_col_partial_kernel(
    const float* __restrict__ S, const int* __restrict__ isNC,
    const float* __restrict__ c, const float* __restrict__ d,
    const float* __restrict__ scal, float* __restrict__ pm, float* __restrict__ ps,
    const int* conv) {
  if (*conv) return;
  const int j = blockIdx.x * 256 + threadIdx.x;
  const int chunk = blockIdx.y;
  if (j >= LDC) return;
  const float invMaxC = scal[1];
  const float dj = d[j];
  float m = -INFINITY, s = 0.f;
  const int r0 = chunk * ROWSPC;
  for (int i = r0; i < r0 + ROWSPC; ++i) {
    __builtin_prefetch(&S[(size_t)(i + 8) * LDC + j], 0, 3);  // 8 rows ahead, same column
    if (mask_ij(i, j, isNC[i])) {
      float val = ((S[(size_t)i * LDC + j] - 1.0f) * invMaxC + c[i] + dj) * kINVREG;
      lse_add(m, s, val);
    }
  }
  pm[(size_t)chunk * LDC + j] = m;
  ps[(size_t)chunk * LDC + j] = s;
}

__global__ __launch_bounds__(256) void sink_col_combine_kernel(
    const float* __restrict__ pm, const float* __restrict__ ps,
    float* __restrict__ d, float logTarget, float* errAcc, const int* conv) {
  if (*conv) return;
  const int j = blockIdx.x * 256 + threadIdx.x;
  if (j >= LDC) return;
  float m = -INFINITY, s = 0.f;
  for (int ch = 0; ch < CHUNKS; ++ch) lse_merge(m, s, pm[(size_t)ch * LDC + j], ps[(size_t)ch * LDC + j]);
  float lse = logf(s + 1e-10f) + m;
  float dj = d[j];
  float dn = kREG * (logTarget - lse) + dj;
  if (errAcc) atomicAdd(errAcc, fabsf(dn - dj));
  d[j] = dn;
}

// emulate the while_loop early stop: after each full iteration, latch convergence
__global__ void sink_check_kernel(float* err, int* conv) {
  if (*conv == 0 && *err < kTHRESH) *conv = 1;
  *err = 0.0f;
}

// ---------------- column stats for the losses: Lcol (col LSE of scale*S) + Rt ------
__global__ __launch_bounds__(256) void colstats_partial_kernel(
    const float* __restrict__ S, const int* __restrict__ isNC,
    const float* __restrict__ u_t, const float* __restrict__ v_t,
    const float* __restrict__ scal,
    float* __restrict__ pmL, float* __restrict__ psL, float* __restrict__ psR) {
  const int j = blockIdx.x * 256 + threadIdx.x;
  const int chunk = blockIdx.y;
  if (j >= LDC) return;
  const float scale = scal[2], invMaxC = scal[1];
  const float utj = u_t[j];
  float m = -INFINITY, s = 0.f, r = 0.f;
  const int r0 = chunk * ROWSPC;
  for (int i = r0; i < r0 + ROWSPC; ++i) {
    __builtin_prefetch(&S[(size_t)(i + 8) * LDC + j], 0, 3);
    float Sij = S[(size_t)i * LDC + j];
    lse_add(m, s, scale * Sij);
    if (mask_ij(i, j, isNC[i]))
      r += __expf(((Sij - 1.0f) * invMaxC + utj + v_t[i]) * kINVREG);
  }
  pmL[(size_t)chunk * LDC + j] = m;
  psL[(size_t)chunk * LDC + j] = s;
  psR[(size_t)chunk * LDC + j] = r;
}

__global__ __launch_bounds__(256) void colstats_combine_kernel(
    const float* __restrict__ pmL, const float* __restrict__ psL, const float* __restrict__ psR,
    float* __restrict__ Lcol, float* __restrict__ Rt) {
  const int j = blockIdx.x * 256 + threadIdx.x;
  if (j >= LDC) return;
  float m = -INFINITY, s = 0.f, r = 0.f;
  for (int ch = 0; ch < CHUNKS; ++ch) {
    lse_merge(m, s, pmL[(size_t)ch * LDC + j], psL[(size_t)ch * LDC + j]);
    r += psR[(size_t)ch * LDC + j];
  }
  Lcol[j] = logf(s) + m;
  Rt[j] = r;
}

// ---------------- final KL accumulation over S (one block per row) ----------------
__global__ __launch_bounds__(256) void final_loss_kernel(
    const float* __restrict__ S, const int* __restrict__ isNC,
    const float* __restrict__ u_v, const float* __restrict__ v_v,
    const float* __restrict__ u_t, const float* __restrict__ v_t,
    const float* __restrict__ Lcol, const float* __restrict__ Rt,
    const float* __restrict__ scal, float* accImg, float* accTxt) {
  const int i = blockIdx.x, tid = threadIdx.x;
  const float scale = scal[2], invMaxC = scal[1];
  const int nci = isNC[i];
  const float uvi = u_v[i], vti = v_t[i];
  const float* row = S + (size_t)i * LDC;

  // pass 1: Li = LSE_j(scale*S) and Rvi = rowsum(pi_v)
  float m = -INFINITY, s = 0.f, rs = 0.f;
  for (int j = tid; j < LDC; j += 256) {
    __builtin_prefetch(row + j + 1024, 0, 3);
    float Sij = row[j];
    lse_add(m, s, scale * Sij);
    if (mask_ij(i, j, nci)) rs += __expf(((Sij - 1.0f) * invMaxC + uvi + v_v[j]) * kINVREG);
  }
  __shared__ float sm[256], ss[256], sr[256];
  sm[tid] = m; ss[tid] = s; sr[tid] = rs; __syncthreads();
  for (int o = 128; o; o >>= 1) {
    if (tid < o) { lse_merge(sm[tid], ss[tid], sm[tid + o], ss[tid + o]); sr[tid] += sr[tid + o]; }
    __syncthreads();
  }
  __shared__ float Li_s, invRvi_s;
  if (tid == 0) { Li_s = logf(ss[0]) + sm[0]; invRvi_s = 1.0f / sr[0]; }
  __syncthreads();
  const float Li = Li_s, invRvi = invRvi_s;

  // pass 2: pointwise KL terms for both directions
  float aI = 0.f, aT = 0.f;
  for (int j = tid; j < LDC; j += 256) {
    float Sij = row[j];
    bool inc = mask_ij(i, j, nci);
    float e = (Sij - 1.0f) * invMaxC;
    float piv = inc ? __expf((e + uvi + v_v[j]) * kINVREG) : 0.f;
    float pit = inc ? __expf((e + u_t[j] + vti) * kINVREG) : 0.f;
    float Lvij = (j == GN) ? (nci ? 1.f : 0.f) : ((i == j && !nci) ? 1.f : 0.f);
    float t  = kGAMMA * piv * invRvi + (1.f - kGAMMA) * Lvij;
    float tt = kGAMMA * pit / Rt[j] + (1.f - kGAMMA) * Lvij;
    float lpI = scale * Sij - Li;
    float lpT = scale * Sij - Lcol[j];
    aI += (t  > 0.f ? t  * logf(t)  : 0.f) - t  * lpI;
    aT += (tt > 0.f ? tt * logf(tt) : 0.f) - tt * lpT;
  }
  __syncthreads();
  sm[tid] = aI; ss[tid] = aT; __syncthreads();
  for (int o = 128; o; o >>= 1) { if (tid < o) { sm[tid] += sm[tid + o]; ss[tid] += ss[tid + o]; } __syncthreads(); }
  if (tid == 0) { atomicAdd(accImg, sm[0]); atomicAdd(accTxt, ss[0]); }
}

__global__ void finalize_kernel(const float* __restrict__ scal, float* __restrict__ out) {
  float loss_img = scal[5] / (float)GN;
  float loss_txt = scal[6] / (float)(GN + 1);
  out[0] = 0.5f * (loss_img + loss_txt);  // loss_ul
  out[1] = scal[7] / (float)GN;           // loss_op
}

// ================================================================================
extern "C" void kernel_launch(void* const* d_in, const int* in_sizes, int n_in,
                              void* d_out, int out_size, void* d_ws, size_t ws_size,
                              hipStream_t stream) {
  const float* image   = (const float*)d_in[0];
  const float* text    = (const float*)d_in[1];
  const float* text_no = (const float*)d_in[2];
  const float* lscale  = (const float*)d_in[3];
  float* out = (float*)d_out;

  // ---- carve workspace (~86 MB assumed available) ----
  size_t off = 0;
  char* base = (char*)d_ws;
  auto carve = [&](size_t bytes) -> void* {
    void* p = base + off;
    off = (off + bytes + 255) & ~(size_t)255;
    return p;
  };
  float*  S    = (float*) carve(sizeof(float) * (size_t)GN * LDC);
  __bf16* Ahi  = (__bf16*)carve(sizeof(__bf16) * (size_t)GN * GD);
  __bf16* Alo  = (__bf16*)carve(sizeof(__bf16) * (size_t)GN * GD);
  __bf16* Bhi  = (__bf16*)carve(sizeof(__bf16) * (size_t)GN * GD);
  __bf16* Blo  = (__bf16*)carve(sizeof(__bf16) * (size_t)GN * GD);
  float*  pm   = (float*) carve(sizeof(float) * (size_t)CHUNKS * LDC);
  float*  ps   = (float*) carve(sizeof(float) * (size_t)CHUNKS * LDC);
  float*  ps2  = (float*) carve(sizeof(float) * (size_t)CHUNKS * LDC);
  float*  u_v  = (float*) carve(sizeof(float) * GN);   // zero-init region starts here
  float*  v_v  = (float*) carve(sizeof(float) * LDC);
  float*  u_t  = (float*) carve(sizeof(float) * LDC);
  float*  v_t  = (float*) carve(sizeof(float) * GN);
  float*  preds= (float*) carve(sizeof(float) * GN);
  float*  Lcol = (float*) carve(sizeof(float) * LDC);
  float*  Rt   = (float*) carve(sizeof(float) * LDC);
  int*    isNC = (int*)   carve(sizeof(int) * GN);
  float*  scal = (float*) carve(sizeof(float) * 64);

  float* err_v  = scal + 3;
  float* err_t  = scal + 4;
  float* accImg = scal + 5;
  float* accTxt = scal + 6;
  float* accOp  = scal + 7;
  int*   conv_v = (int*)(scal + 8);
  int*   conv_t = (int*)(scal + 9);

  const size_t zero_ints = ((base + off) - (char*)u_v) / sizeof(int);
  const float logN  = logf((float)GN);
  const float logN1 = logf((float)(GN + 1));

  // ---- init ----
  zero_kernel<<<256, 256, 0, stream>>>((int*)u_v, zero_ints);
  setmin_kernel<<<1, 1, 0, stream>>>((unsigned*)scal);

  // ---- bf16 split conversion ----
  cvt_split_kernel<<<(GN * GD + 255) / 256, 256, 0, stream>>>(image, Ahi, Alo, GN * GD);
  cvt_split_kernel<<<(GN * GD + 255) / 256, 256, 0, stream>>>(text,  Bhi, Blo, GN * GD);

  // ---- S = [image @ text^T | rowdot(image, text_no)] ----
  gemm_wmma_kernel<<<dim3(GN / 32, GN / 64), 256, 0, stream>>>(Ahi, Alo, Bhi, Blo, S);
  rowdot_kernel<<<GN / 8, 256, 0, stream>>>(image, text_no, S);

  // ---- loss_op (independent) ----
  loss_op_kernel<<<GN / 8, 256, 0, stream>>>(text, text_no, accOp);

  // ---- cost normalization + scale ----
  gmin_kernel<<<1024, 256, 0, stream>>>(S, (unsigned*)scal);
  prep_kernel<<<1, 1, 0, stream>>>(lscale, scal);

  // ---- preds + top-k smallest -> isNC ----
  preds_kernel<<<GN, 256, 0, stream>>>(S, scal, preds);
  topk_kernel<<<1, 1024, 0, stream>>>(preds, isNC);

  const dim3 colGrid((LDC + 255) / 256, CHUNKS);
  const int  combGrid = (LDC + 255) / 256;

  // ---- Sinkhorn V: rows=n, cols=n+1; p=1/n, q=1/(n+1) ----
  for (int it = 0; it < 100; ++it) {
    sink_row_kernel<<<GN, 256, 0, stream>>>(S, isNC, v_v, u_v, scal, -logN, err_v, conv_v);
    sink_col_partial_kernel<<<colGrid, 256, 0, stream>>>(S, isNC, u_v, v_v, scal, pm, ps, conv_v);
    sink_col_combine_kernel<<<combGrid, 256, 0, stream>>>(pm, ps, v_v, -logN1, nullptr, conv_v);
    sink_check_kernel<<<1, 1, 0, stream>>>(err_v, conv_v);
  }

  // ---- Sinkhorn T (transposed problem): u_t over n+1 (col pass), v_t over n (row pass) ----
  for (int it = 0; it < 100; ++it) {
    sink_col_partial_kernel<<<colGrid, 256, 0, stream>>>(S, isNC, v_t, u_t, scal, pm, ps, conv_t);
    sink_col_combine_kernel<<<combGrid, 256, 0, stream>>>(pm, ps, u_t, -logN1, err_t, conv_t);
    sink_row_kernel<<<GN, 256, 0, stream>>>(S, isNC, u_t, v_t, scal, -logN, nullptr, conv_t);
    sink_check_kernel<<<1, 1, 0, stream>>>(err_t, conv_t);
  }

  // ---- column stats (Lcol for log_softmax of logits^T, Rt = rowsums of pi_t) ----
  colstats_partial_kernel<<<colGrid, 256, 0, stream>>>(S, isNC, u_t, v_t, scal, pm, ps, ps2);
  colstats_combine_kernel<<<combGrid, 256, 0, stream>>>(pm, ps, ps2, Lcol, Rt);

  // ---- KL losses + finalize ----
  final_loss_kernel<<<GN, 256, 0, stream>>>(S, isNC, u_v, v_v, u_t, v_t, Lcol, Rt, scal, accImg, accTxt);
  finalize_kernel<<<1, 1, 0, stream>>>(scal, out);

  (void)in_sizes; (void)n_in; (void)out_size; (void)ws_size;
}
